// MoEDecoderLayer_61400852464238
// MI455X (gfx1250) — matrix operand order repaired and checked by reading, not compiled, not run on hardware
//
#include <hip/hip_runtime.h>
#include <hip/hip_bf16.h>

#define T_TOK 4096
#define Hdim  2048
#define Idim  1024
#define Edim  32
#define Ktop  4
#define CAP   1024      // 2*T*K/E, same capacity as reference
#define EPSV  1e-6f

#define BM 128
#define BN 64
#define BK 32
#define NTHREADS 256

#if __has_builtin(__builtin_amdgcn_global_load_async_to_lds_b128) && \
    __has_builtin(__builtin_amdgcn_s_wait_asynccnt)
#define HAVE_ASYNC_LDS 1
#else
#define HAVE_ASYNC_LDS 0
#endif

typedef __attribute__((ext_vector_type(16))) __bf16 bf16x16;
typedef __attribute__((ext_vector_type(8)))  __bf16 bf16x8;
typedef __attribute__((ext_vector_type(2)))  __bf16 bf16x2;
typedef __attribute__((ext_vector_type(8)))  float  f32x8;

#if HAVE_ASYNC_LDS
typedef __attribute__((ext_vector_type(4))) int v4i;
typedef __attribute__((address_space(1))) v4i gv4i;   // global-qualified v4i
typedef __attribute__((address_space(3))) v4i lv4i;   // LDS-qualified v4i
// async DMA 16 bytes global -> LDS (ASYNCcnt-tracked)
__device__ __forceinline__ void async_b128(const __bf16* g, __bf16* l) {
  __builtin_amdgcn_global_load_async_to_lds_b128((gv4i*)g, (lv4i*)l, 0, 0);
}
#endif

// ---------------- RMSNorm + residual init ----------------
__global__ __launch_bounds__(NTHREADS)
void rmsnorm_residual_kernel(const float* __restrict__ x,
                             const float* __restrict__ w,
                             __bf16* __restrict__ xb,
                             float* __restrict__ out) {
  int t = blockIdx.x;
  const float* row = x + (size_t)t * Hdim;
  __shared__ float red[8];
  float ss = 0.f;
  for (int i = threadIdx.x; i < Hdim; i += NTHREADS) {
    float v = row[i];
    ss += v * v;
  }
  for (int o = 16; o > 0; o >>= 1) ss += __shfl_xor(ss, o, 32);
  if ((threadIdx.x & 31) == 0) red[threadIdx.x >> 5] = ss;
  __syncthreads();
  if (threadIdx.x < 8) {
    float v = red[threadIdx.x];
    for (int o = 4; o > 0; o >>= 1) v += __shfl_xor(v, o, 8);
    if (threadIdx.x == 0) red[0] = v;
  }
  __syncthreads();
  float inv = rsqrtf(red[0] / (float)Hdim + EPSV);
  for (int i = threadIdx.x; i < Hdim; i += NTHREADS) {
    float v = row[i];
    out[(size_t)t * Hdim + i] = v;                    // residual
    xb[(size_t)t * Hdim + i] = (__bf16)(v * inv * w[i]);
  }
}

// ---------------- Gate: logits -> softmax -> top4 -> renorm ----------------
__global__ __launch_bounds__(NTHREADS)
void gate_topk_kernel(const __bf16* __restrict__ xb,
                      const float* __restrict__ gw,
                      int* __restrict__ topk_ids,
                      float* __restrict__ topk_w) {
  int t = blockIdx.x;
  __shared__ float xs[Hdim];
  __shared__ float logits_s[Edim];
  for (int i = threadIdx.x; i < Hdim; i += NTHREADS)
    xs[i] = (float)xb[(size_t)t * Hdim + i];
  __syncthreads();

  int wid = threadIdx.x >> 5, lane = threadIdx.x & 31;
  int e = wid * 4 + (lane >> 3);   // 8 lanes per expert, 4 experts per wave
  int sub = lane & 7;
  const float* wrow = gw + (size_t)e * Hdim;
  float acc = 0.f;
  for (int i = sub; i < Hdim; i += 8) acc += xs[i] * wrow[i];
  for (int o = 4; o > 0; o >>= 1) acc += __shfl_xor(acc, o, 32);
  if (sub == 0) logits_s[e] = acc;
  __syncthreads();

  if (wid == 0) {
    float lg = logits_s[lane];
    float mx = lg;
    for (int o = 16; o > 0; o >>= 1) mx = fmaxf(mx, __shfl_xor(mx, o, 32));
    float ex = __expf(lg - mx);
    float sm = ex;
    for (int o = 16; o > 0; o >>= 1) sm += __shfl_xor(sm, o, 32);
    float p = ex * __builtin_amdgcn_rcpf(sm);
    float topv[Ktop];
    int topi[Ktop];
    float cur = p;
#pragma unroll
    for (int k = 0; k < Ktop; k++) {
      float mv = cur; int mi = lane;
      for (int o = 16; o > 0; o >>= 1) {
        float ov = __shfl_xor(mv, o, 32);
        int   oi = __shfl_xor(mi, o, 32);
        if (ov > mv || (ov == mv && oi < mi)) { mv = ov; mi = oi; }
      }
      topv[k] = mv; topi[k] = mi;
      if (lane == mi) cur = -1e30f;
    }
    float rws = __builtin_amdgcn_rcpf(topv[0] + topv[1] + topv[2] + topv[3]);
    if (lane == 0) {
#pragma unroll
      for (int k = 0; k < Ktop; k++) {
        topk_ids[t * Ktop + k] = topi[k];
        topk_w[t * Ktop + k]   = topv[k] * rws;
      }
    }
  }
}

// ---------------- Dispatch ----------------
__global__ void zero_counts_kernel(int* counts) {
  if (threadIdx.x < Edim) counts[threadIdx.x] = 0;
}

__global__ void dispatch_kernel(const int* __restrict__ topk_ids,
                                const float* __restrict__ topk_w,
                                int* __restrict__ counts,
                                int* __restrict__ slot_tok,
                                float* __restrict__ slot_w) {
  int g = blockIdx.x * blockDim.x + threadIdx.x;
  if (g >= T_TOK * Ktop) return;
  int e = topk_ids[g];
  float w = topk_w[g];
  int pos = atomicAdd(&counts[e], 1);
  if (pos < CAP) {
    slot_tok[e * CAP + pos] = g / Ktop;
    slot_w[e * CAP + pos]   = w;
  }
}

// ---------------- Fragment helpers ----------------
// A 16x32 bf16 fragment: lane%16 = row, lane/16 selects contiguous K half (k = 16*(lane/16) + s).
__device__ __forceinline__ bf16x16 load_afrag(const __bf16* As, int rowbase, int lane) {
  int row = rowbase + (lane & 15);
  int c0 = 16 * (lane >> 4);
  bf16x8 lo = *(const bf16x8*)(&As[row * BK + c0]);
  bf16x8 hi = *(const bf16x8*)(&As[row * BK + c0 + 8]);
  bf16x16 a;
#pragma unroll
  for (int i = 0; i < 8; i++) { a[i] = lo[i]; a[8 + i] = hi[i]; }
  return a;
}

// B LDS layout (per 16-col fragment f): d = f*512 + lane*16 + e holds
// B[k = 16*(lane/16) + e][n = f*16 + lane%16]  -> lane's 16 values contiguous.
// Staging map: wave w covers (f = w&3, khalf = w>>2); each 16-lane group reads a
// 64B contiguous row segment of B (coalesced); thread keeps 8 consecutive-k
// values of one column in regs, packs pairs with v_cvt_pk_bf16_f32.
__device__ __forceinline__ void stage_b_idx(int tid, int nb, int ldb,
                                            int* goff, int* lds) {
  int lane = tid & 31, w = tid >> 5;
  int f = w & 3, half = w >> 2;
  int k = ((lane >> 4) << 4) + half * 8;
  int n = (f << 4) + (lane & 15);
  *goff = k * ldb + nb + n;
  *lds  = f * 512 + lane * 16 + half * 8;
}

template <int LDB>
__device__ __forceinline__ void fetch_b(const float* __restrict__ Wk, int goff,
                                        float r[8]) {
#pragma unroll
  for (int i = 0; i < 8; i++) r[i] = Wk[goff + i * LDB];
}

__device__ __forceinline__ void commit_b(__bf16* Bs, int lds, const float r[8]) {
#pragma unroll
  for (int i = 0; i < 8; i += 2) {
    bf16x2 p;
    p[0] = (__bf16)r[i];
    p[1] = (__bf16)r[i + 1];
    *(bf16x2*)(&Bs[lds + i]) = p;
  }
}

// ---------------- GEMM 1: x @ [Wg|Wu] -> SwiGLU -> bf16 act ----------------
template <bool ROUTED, int LDB, int KDIM>
__global__ __launch_bounds__(NTHREADS)
void gemm_swiglu_kernel(const __bf16* __restrict__ A,       // [*, KDIM] bf16
                        const int* __restrict__ slot_tok,
                        const int* __restrict__ counts,
                        const float* __restrict__ Wg_base,
                        const float* __restrict__ Wu_base,
                        long expert_stride,
                        __bf16* __restrict__ actOut, int Mfixed) {
  int e = blockIdx.z;
  int mtile = blockIdx.y;
  int nb = blockIdx.x * BN;
  int M, rowbase;
  if constexpr (ROUTED) {
    int c = counts[e];
    M = c < CAP ? c : CAP;
    rowbase = e * CAP;
  } else {
    M = Mfixed;
    rowbase = 0;
  }
  if (mtile * BM >= M) return;

  const float* Wg = Wg_base + (size_t)e * expert_stride;
  const float* Wu = Wu_base + (size_t)e * expert_stride;

  __shared__ __attribute__((aligned(16))) __bf16 As[2][BM * BK];
  __shared__ __attribute__((aligned(16))) __bf16 Bgs[2][BK * BN];
  __shared__ __attribute__((aligned(16))) __bf16 Bus[2][BK * BN];

  int tid = threadIdx.x, wid = tid >> 5, lane = tid & 31;
  int waveM = wid & 3, waveN = wid >> 2;

  f32x8 accG[2][2] = {};
  f32x8 accU[2][2] = {};

  // A staging: 2 threads per row, 16 bf16 (32B) each
  int arow = tid >> 1;
  int acol = (tid & 1) * 16;
  int gm = mtile * BM + arow;
  long a_src;
  if constexpr (ROUTED) {
    int tok = (gm < M) ? slot_tok[e * CAP + gm] : 0;
    a_src = (long)tok * KDIM;
  } else {
    a_src = (long)gm * KDIM;
  }
  const __bf16* aG = A + a_src + acol;       // + kb
  __bf16* aL0 = &As[0][arow * BK + acol];
  __bf16* aL1 = &As[1][arow * BK + acol];

  int goff, blds;
  stage_b_idx(tid, nb, LDB, &goff, &blds);

  float bg_r[8], bu_r[8];
#if !HAVE_ASYNC_LDS
  uint4 a_r[2];
#endif

  // prefetch tile 0
#if HAVE_ASYNC_LDS
  async_b128(aG, aL0);
  async_b128(aG + 8, aL0 + 8);
#else
  {
    const uint4* src = (const uint4*)aG;
    a_r[0] = src[0];
    a_r[1] = src[1];
  }
#endif
  fetch_b<LDB>(Wg, goff, bg_r);
  fetch_b<LDB>(Wu, goff, bu_r);

  int buf = 0;
  for (int kb = 0; kb < KDIM; kb += BK) {
    // commit staged B tile into LDS[buf]
    commit_b(Bgs[buf], blds, bg_r);
    commit_b(Bus[buf], blds, bu_r);
#if HAVE_ASYNC_LDS
    __builtin_amdgcn_s_wait_asynccnt(0);   // A[buf] DMA complete
#else
    *(uint4*)(&As[buf][arow * BK + acol])     = a_r[0];
    *(uint4*)(&As[buf][arow * BK + acol + 8]) = a_r[1];
#endif
    __syncthreads();

    // prefetch next tile (overlaps with the WMMAs below)
    if (kb + BK < KDIM) {
#if HAVE_ASYNC_LDS
      __bf16* dst = buf ? aL0 : aL1;
      async_b128(aG + kb + BK, dst);
      async_b128(aG + kb + BK + 8, dst + 8);
#else
      const uint4* src = (const uint4*)(aG + kb + BK);
      a_r[0] = src[0];
      a_r[1] = src[1];
#endif
      fetch_b<LDB>(Wg + (size_t)(kb + BK) * LDB, goff, bg_r);
      fetch_b<LDB>(Wu + (size_t)(kb + BK) * LDB, goff, bu_r);
    }

    bf16x16 afrag[2], bg[2], bu[2];
#pragma unroll
    for (int fm = 0; fm < 2; fm++)
      afrag[fm] = load_afrag(As[buf], waveM * 32 + fm * 16, lane);
#pragma unroll
    for (int fn = 0; fn < 2; fn++) {
      int f = waveN * 2 + fn;
      bg[fn] = *(const bf16x16*)(&Bgs[buf][f * 512 + lane * 16]);
      bu[fn] = *(const bf16x16*)(&Bus[buf][f * 512 + lane * 16]);
    }
#pragma unroll
    for (int fm = 0; fm < 2; fm++)
#pragma unroll
      for (int fn = 0; fn < 2; fn++) {
        accG[fm][fn] = __builtin_amdgcn_wmma_f32_16x16x32_bf16(
            false, afrag[fm], false, bg[fn], (short)0, accG[fm][fn], false, false);
        accU[fm][fn] = __builtin_amdgcn_wmma_f32_16x16x32_bf16(
            false, afrag[fm], false, bu[fn], (short)0, accU[fm][fn], false, false);
      }
    buf ^= 1;
  }

#pragma unroll
  for (int fm = 0; fm < 2; fm++)
#pragma unroll
    for (int fn = 0; fn < 2; fn++)
#pragma unroll
      for (int r = 0; r < 8; r++) {
        int m = mtile * BM + waveM * 32 + fm * 16 + (lane >> 4) * 8 + r;
        if (m >= M) continue;
        int col = nb + waveN * 32 + fn * 16 + (lane & 15);
        float g = accG[fm][fn][r];
        float u = accU[fm][fn][r];
        // silu(g)*u with fast rcp (v_rcp_f32) instead of IEEE divide
        float s = g * __builtin_amdgcn_rcpf(1.0f + __expf(-g)) * u;
        actOut[(size_t)(rowbase + m) * Idim + col] = (__bf16)s;
      }
}

// ---------------- GEMM 2: act @ W2 -> weighted scatter-add into out ----------------
template <bool ROUTED, int LDB, int KDIM>
__global__ __launch_bounds__(NTHREADS)
void gemm_down_kernel(const __bf16* __restrict__ act,
                      const int* __restrict__ slot_tok,
                      const float* __restrict__ slot_w,
                      const int* __restrict__ counts,
                      const float* __restrict__ W_base,
                      long expert_stride,
                      float* __restrict__ out, int Mfixed) {
  int e = blockIdx.z;
  int mtile = blockIdx.y;
  int nb = blockIdx.x * BN;
  int M, rowbase;
  if constexpr (ROUTED) {
    int c = counts[e];
    M = c < CAP ? c : CAP;
    rowbase = e * CAP;
  } else {
    M = Mfixed;
    rowbase = 0;
  }
  if (mtile * BM >= M) return;

  const float* W = W_base + (size_t)e * expert_stride;

  __shared__ __attribute__((aligned(16))) __bf16 As[2][BM * BK];
  __shared__ __attribute__((aligned(16))) __bf16 Bs[2][BK * BN];

  int tid = threadIdx.x, wid = tid >> 5, lane = tid & 31;
  int waveM = wid & 3, waveN = wid >> 2;
  f32x8 acc[2][2] = {};

  int arow = tid >> 1;
  int acol = (tid & 1) * 16;
  long a_src = (long)(rowbase + mtile * BM + arow) * KDIM;
  const __bf16* aG = act + a_src + acol;
  __bf16* aL0 = &As[0][arow * BK + acol];
  __bf16* aL1 = &As[1][arow * BK + acol];

  int goff, blds;
  stage_b_idx(tid, nb, LDB, &goff, &blds);

  float b_r[8];
#if !HAVE_ASYNC_LDS
  uint4 a_r[2];
#endif

#if HAVE_ASYNC_LDS
  async_b128(aG, aL0);
  async_b128(aG + 8, aL0 + 8);
#else
  {
    const uint4* src = (const uint4*)aG;
    a_r[0] = src[0];
    a_r[1] = src[1];
  }
#endif
  fetch_b<LDB>(W, goff, b_r);

  int buf = 0;
  for (int kb = 0; kb < KDIM; kb += BK) {
    commit_b(Bs[buf], blds, b_r);
#if HAVE_ASYNC_LDS
    __builtin_amdgcn_s_wait_asynccnt(0);
#else
    *(uint4*)(&As[buf][arow * BK + acol])     = a_r[0];
    *(uint4*)(&As[buf][arow * BK + acol + 8]) = a_r[1];
#endif
    __syncthreads();

    if (kb + BK < KDIM) {
#if HAVE_ASYNC_LDS
      __bf16* dst = buf ? aL0 : aL1;
      async_b128(aG + kb + BK, dst);
      async_b128(aG + kb + BK + 8, dst + 8);
#else
      const uint4* src = (const uint4*)(aG + kb + BK);
      a_r[0] = src[0];
      a_r[1] = src[1];
#endif
      fetch_b<LDB>(W + (size_t)(kb + BK) * LDB, goff, b_r);
    }

    bf16x16 afrag[2], bf[2];
#pragma unroll
    for (int fm = 0; fm < 2; fm++)
      afrag[fm] = load_afrag(As[buf], waveM * 32 + fm * 16, lane);
#pragma unroll
    for (int fn = 0; fn < 2; fn++)
      bf[fn] = *(const bf16x16*)(&Bs[buf][(waveN * 2 + fn) * 512 + lane * 16]);
#pragma unroll
    for (int fm = 0; fm < 2; fm++)
#pragma unroll
      for (int fn = 0; fn < 2; fn++)
        acc[fm][fn] = __builtin_amdgcn_wmma_f32_16x16x32_bf16(
            false, afrag[fm], false, bf[fn], (short)0, acc[fm][fn], false, false);
    buf ^= 1;
  }

#pragma unroll
  for (int fm = 0; fm < 2; fm++)
#pragma unroll
    for (int fn = 0; fn < 2; fn++)
#pragma unroll
      for (int r = 0; r < 8; r++) {
        int m = mtile * BM + waveM * 32 + fm * 16 + (lane >> 4) * 8 + r;
        if (m >= M) continue;
        int col = nb + waveN * 32 + fn * 16 + (lane & 15);
        int tok; float wgt;
        if constexpr (ROUTED) {
          tok = slot_tok[e * CAP + m];
          wgt = slot_w[e * CAP + m];
        } else {
          tok = m;
          wgt = 1.0f;
        }
        atomicAdd(&out[(size_t)tok * Hdim + col], wgt * acc[fm][fn][r]);
      }
}

// ---------------- Launch ----------------
extern "C" void kernel_launch(void* const* d_in, const int* in_sizes, int n_in,
                              void* d_out, int out_size, void* d_ws, size_t ws_size,
                              hipStream_t stream) {
  const float* x   = (const float*)d_in[0];
  const float* nw  = (const float*)d_in[1];
  const float* gw  = (const float*)d_in[2];
  const float* w13 = (const float*)d_in[3];
  const float* w2  = (const float*)d_in[4];
  const float* sgw = (const float*)d_in[5];
  const float* suw = (const float*)d_in[6];
  const float* sdw = (const float*)d_in[7];
  float* out = (float*)d_out;

  char* ws = (char*)d_ws;
  size_t off = 0;
  auto carve = [&](size_t bytes) -> char* {
    char* p = ws + off;
    off = (off + bytes + 255) & ~(size_t)255;
    return p;
  };
  __bf16* xb       = (__bf16*)carve((size_t)T_TOK * Hdim * 2);
  __bf16* act_r    = (__bf16*)carve((size_t)Edim * CAP * Idim * 2);
  __bf16* act_s    = (__bf16*)carve((size_t)T_TOK * Idim * 2);
  int*    topk_ids = (int*)carve((size_t)T_TOK * Ktop * 4);
  float*  topk_w   = (float*)carve((size_t)T_TOK * Ktop * 4);
  int*    counts   = (int*)carve(Edim * 4);
  int*    slot_tok = (int*)carve((size_t)Edim * CAP * 4);
  float*  slot_w   = (float*)carve((size_t)Edim * CAP * 4);

  rmsnorm_residual_kernel<<<T_TOK, NTHREADS, 0, stream>>>(x, nw, xb, out);
  gate_topk_kernel<<<T_TOK, NTHREADS, 0, stream>>>(xb, gw, topk_ids, topk_w);
  zero_counts_kernel<<<1, 64, 0, stream>>>(counts);
  dispatch_kernel<<<(T_TOK * Ktop + 255) / 256, 256, 0, stream>>>(
      topk_ids, topk_w, counts, slot_tok, slot_w);

  // routed experts: gu = buf @ w13 ; act = silu(g)*u   (ldb = 2I, K = H)
  dim3 g1(Idim / BN, CAP / BM, Edim);
  gemm_swiglu_kernel<true, 2 * Idim, Hdim><<<g1, NTHREADS, 0, stream>>>(
      xb, slot_tok, counts, w13, w13 + Idim, (long)Hdim * 2 * Idim, act_r, 0);
  // routed experts: out += w * (act @ w2)              (ldb = H, K = I)
  dim3 g2(Hdim / BN, CAP / BM, Edim);
  gemm_down_kernel<true, Hdim, Idim><<<g2, NTHREADS, 0, stream>>>(
      act_r, slot_tok, slot_w, counts, w2, (long)Idim * Hdim, out, 0);

  // shared expert                                      (ldb = I, K = H)
  dim3 g3(Idim / BN, T_TOK / BM, 1);
  gemm_swiglu_kernel<false, Idim, Hdim><<<g3, NTHREADS, 0, stream>>>(
      xb, nullptr, nullptr, sgw, suw, 0, act_s, T_TOK);
  dim3 g4(Hdim / BN, T_TOK / BM, 1);
  gemm_down_kernel<false, Hdim, Idim><<<g4, NTHREADS, 0, stream>>>(
      act_s, nullptr, nullptr, nullptr, sdw, 0, out, T_TOK);
}
